// HyperformerPairBlock_23218593202744
// MI455X (gfx1250) — matrix operand order repaired and checked by reading, not compiled, not run on hardware
//
#include <hip/hip_runtime.h>
#include <hip/hip_bf16.h>

typedef __attribute__((ext_vector_type(2)))  float        v2f;
typedef __attribute__((ext_vector_type(8)))  float        v8f;
typedef __attribute__((ext_vector_type(16))) __bf16       v16bf;
typedef __attribute__((ext_vector_type(4)))  unsigned int u32x4;
typedef __attribute__((ext_vector_type(8)))  int          i32x8;
typedef __attribute__((ext_vector_type(4)))  int          i32x4;

#define A_N 384
#define F_N 128
#define C_N 32
#define H_N 512
#define EPSN 1e-5f

// ---------------------------------------------------------------------------
// TDM helpers: build D# (ISA cdna5 §8.3/8.4) for a 2D tile of 8-byte elements.
//   group0: count=1, lds_addr, global_addr[56:0], type=2
//   group1: data_size=3 (8B), tensor_dim0/1, tile_dim0/1, tensor_dim0_stride
// groups 2/3 zero (<=2D tensor). 6-arg builtin form (clang-23 toolchain).
// ---------------------------------------------------------------------------
__device__ __forceinline__ void tdm_build(u32x4& g0, i32x8& g1,
                                          unsigned lds_off, unsigned long long ga,
                                          unsigned tile0, unsigned tile1,
                                          unsigned td0, unsigned td1,
                                          unsigned long long str0) {
  g0[0] = 1u;                                   // count=1, is_restore=0, gather off
  g0[1] = lds_off;                              // LDS byte address
  g0[2] = (unsigned)ga;                         // global addr lo
  g0[3] = (unsigned)((ga >> 32) & 0x1FFFFFFull) | (2u << 30);  // addr hi | type=2
  g1[0] = (int)(3u << 16);                      // data_size = 8 bytes
  g1[1] = (int)((td0 & 0xFFFFu) << 16);         // tensor_dim0[15:0]
  g1[2] = (int)(((td0 >> 16) & 0xFFFFu) | ((td1 & 0xFFFFu) << 16));
  g1[3] = (int)(((td1 >> 16) & 0xFFFFu) | ((tile0 & 0xFFFFu) << 16));
  g1[4] = (int)(tile1 & 0xFFFFu);               // tile_dim1 | tile_dim2=0
  g1[5] = (int)(unsigned)(str0 & 0xFFFFFFFFull);
  g1[6] = (int)(unsigned)((str0 >> 32) & 0xFFFFull);
  g1[7] = 0;                                    // tensor_dim1_stride unused (2D)
}
__device__ __forceinline__ void tdm_load_2d(unsigned lds_off, const void* gp,
                                            unsigned tile0, unsigned tile1,
                                            unsigned td0, unsigned td1,
                                            unsigned long long str0) {
  u32x4 g0; i32x8 g1;
  tdm_build(g0, g1, lds_off, (unsigned long long)(size_t)gp, tile0, tile1, td0, td1, str0);
  i32x4 z4; z4[0] = z4[1] = z4[2] = z4[3] = 0;
  i32x8 z8; for (int i = 0; i < 8; ++i) z8[i] = 0;
  __builtin_amdgcn_tensor_load_to_lds(g0, g1, z4, z4, z8, 0);
}
__device__ __forceinline__ void tdm_store_2d(unsigned lds_off, void* gp,
                                             unsigned tile0, unsigned tile1,
                                             unsigned td0, unsigned td1,
                                             unsigned long long str0) {
  u32x4 g0; i32x8 g1;
  tdm_build(g0, g1, lds_off, (unsigned long long)(size_t)gp, tile0, tile1, td0, td1, str0);
  i32x4 z4; z4[0] = z4[1] = z4[2] = z4[3] = 0;
  i32x8 z8; for (int i = 0; i < 8; ++i) z8[i] = 0;
  __builtin_amdgcn_tensor_store_from_lds(g0, g1, z4, z4, z8, 0);
}

// ---------------------------------------------------------------------------
// P1: act = LayerNorm(node_vec)*s+b ; left/right = (act@W +b)*mask  [384x32]
// ---------------------------------------------------------------------------
__global__ void prep_act_lr(const float* __restrict__ node_vec,
                            const float* __restrict__ node_mask,
                            const float* __restrict__ opS, const float* __restrict__ opB,
                            const float* __restrict__ wL, const float* __restrict__ bL,
                            const float* __restrict__ wR, const float* __restrict__ bR,
                            float* __restrict__ left, float* __restrict__ right) {
  const int a = blockIdx.x, t = threadIdx.x;
  __shared__ float sh[F_N];
  __shared__ float red[128];
  float x = node_vec[a * F_N + t];
  red[t] = x; __syncthreads();
  for (int s = 64; s > 0; s >>= 1) { if (t < s) red[t] += red[t + s]; __syncthreads(); }
  float mu = red[0] * (1.0f / F_N); __syncthreads();
  float d = x - mu;
  red[t] = d * d; __syncthreads();
  for (int s = 64; s > 0; s >>= 1) { if (t < s) red[t] += red[t + s]; __syncthreads(); }
  float rs = rsqrtf(red[0] * (1.0f / F_N) + EPSN);
  sh[t] = d * rs * opS[t] + opB[t];
  __syncthreads();
  if (t < C_N) {
    float m = node_mask[a];
    float aL = bL[t], aR = bR[t];
    for (int f = 0; f < F_N; ++f) { float v = sh[f]; aL += v * wL[f * C_N + t]; aR += v * wR[f * C_N + t]; }
    left[a * C_N + t] = aL * m;
    right[a * C_N + t] = aR * m;
  }
}

// ---------------------------------------------------------------------------
// P2: M[b,c,f] = sum_d right[b,d]*w_out[(c*32+d),f] in f32 16x16x4 B-frag
// layout: MB4[b][nt(8)][kt(8)][lane(32)][v(2)]
// ---------------------------------------------------------------------------
__global__ void prep_MB4(const float* __restrict__ right, const float* __restrict__ w_out,
                         float* __restrict__ MB4) {
  int i = blockIdx.x * 256 + threadIdx.x;  // 1572864
  int v = i & 1, lane = (i >> 1) & 31, kt = (i >> 6) & 7, nt = (i >> 9) & 7, b = i >> 12;
  int c = kt * 4 + (lane >> 4) * 2 + v;
  int f = nt * 16 + (lane & 15);
  float s = 0.f;
  for (int d = 0; d < C_N; ++d) s += right[b * C_N + d] * w_out[(c * C_N + d) * F_N + f];
  MB4[i] = s;
}

// P3: leftA4[at(24)][kt(8)][lane(32)][v(2)] f32 A-frag (16x4)
__global__ void prep_leftA4(const float* __restrict__ left, float* __restrict__ leftA4) {
  int i = blockIdx.x * 256 + threadIdx.x;  // 12288
  int v = i & 1, lane = (i >> 1) & 31, kt = (i >> 6) & 7, at = i >> 9;
  int a = at * 16 + (lane & 15);
  int c = kt * 4 + (lane >> 4) * 2 + v;
  leftA4[i] = left[a * C_N + c];
}

// P4/P5: bf16 32x16 B-fragments for w_t1 / w_t2
__global__ void prep_w1B(const float* __restrict__ w1, __bf16* __restrict__ w1B) {
  int i = blockIdx.x * 256 + threadIdx.x;  // 65536
  int e = i & 15, lane = (i >> 4) & 31, nt = (i >> 9) & 31, kt = i >> 14;
  int k = kt * 32 + (lane >> 4) * 16 + e;
  int n = nt * 16 + (lane & 15);
  w1B[i] = (__bf16)w1[k * H_N + n];
}
__global__ void prep_w2B(const float* __restrict__ w2, __bf16* __restrict__ w2B) {
  int i = blockIdx.x * 256 + threadIdx.x;  // 65536
  int e = i & 15, lane = (i >> 4) & 31, nt = (i >> 9) & 7, kt = i >> 12;
  int k = kt * 32 + (lane >> 4) * 16 + e;
  int n = nt * 16 + (lane & 15);
  w2B[i] = (__bf16)w2[k * F_N + n];
}

// ---------------------------------------------------------------------------
// LDS pool layout (byte offsets) -- single pool so TDM lds_addr is known
// ---------------------------------------------------------------------------
#define OFF_W1S 0                         // 131072 B : w1 bf16 frags
#define OFF_W2S 131072                    // 131072 B : w2 bf16 frags
#define OFF_ELD 262144                    //  16384 B : edge tile double buffer
#define OFF_ESM 278528                    //   8192 B : f32 work tile
#define OFF_T1S 286720                    //  16384 B : relu(t1) bf16, per wave
#define OFF_PRM 303104                    //   4096 B : lnS,lnB,bOut,bT2,bT1
#define OFF_RED 307200                    //   1152 B : LN partials
#define POOL_SZ 308352

__launch_bounds__(128, 1)
__global__ void pair_block_main(const float* __restrict__ edge_vec,
                                const float* __restrict__ b_out,
                                const float* __restrict__ trS, const float* __restrict__ trB,
                                const float* __restrict__ bT1, const float* __restrict__ bT2,
                                const float* __restrict__ leftA4, const float* __restrict__ MB4,
                                const __bf16* __restrict__ w1B, const __bf16* __restrict__ w2B,
                                float* __restrict__ out) {
  __shared__ __align__(32) char pool[POOL_SZ];
  __bf16* w1s = (__bf16*)(pool + OFF_W1S);
  __bf16* w2s = (__bf16*)(pool + OFF_W2S);
  float*  Eld = (float*)(pool + OFF_ELD);                 // [2][16][128]
  float (*Esm)[128] = (float(*)[128])(pool + OFF_ESM);    // [16][128]
  __bf16* t1p = (__bf16*)(pool + OFF_T1S);                // [4][16][128]
  float* lnS = (float*)(pool + OFF_PRM);
  float* lnB = lnS + 128; float* bOutS = lnB + 128;
  float* bT2s = bOutS + 128; float* bT1s = bT2s + 128;    // bT1s: 512
  float (*pS)[8] = (float(*)[8])(pool + OFF_RED);
  float (*pQ)[8] = pS + 16;
  float* muS = (float*)(pQ + 16);
  float* rsS = muS + 16;

  const int tid = threadIdx.x;
  const int w = tid >> 5, lane = tid & 31;
  const int row = lane & 15, kh = lane >> 4;
  const int at = blockIdx.y;   // 0..23
  const int bg = blockIdx.x;   // 0..15
  const int a0 = at * 16;
  const unsigned long long ROWSTR8 = (unsigned long long)A_N * F_N / 2;  // 24576 (8B units)

  // ---- params via plain loads; weights + first tile via TDM (wave 0) ----
  lnS[tid] = trS[tid]; lnB[tid] = trB[tid];
  bOutS[tid] = b_out[tid]; bT2s[tid] = bT2[tid];
  for (int i = tid; i < H_N; i += 128) bT1s[i] = bT1[i];
  if (w == 0) {
    tdm_load_2d(OFF_W1S, w1B, 16384, 1, 16384, 1, 16384ull);
    tdm_load_2d(OFF_W2S, w2B, 16384, 1, 16384, 1, 16384ull);
    tdm_load_2d(OFF_ELD, edge_vec + ((size_t)a0 * A_N + bg * 24) * F_N,
                64, 16, 64, 16, ROWSTR8);
  }
  // left A-fragments (f32 16x16x4), reused for all b
  v2f la[8];
  for (int kt = 0; kt < 8; ++kt)
    la[kt] = *(const v2f*)&leftA4[((at * 8 + kt) * 32 + lane) * 2];
  if (w == 0) __builtin_amdgcn_s_wait_tensorcnt(0);
  __syncthreads();

  const int cr = tid >> 3, cc = tid & 7;  // cooperative row / 16-col chunk

  for (int bi = 0; bi < 24; ++bi) {
    const int b = bg * 24 + bi;
    const int buf = bi & 1;
    if (bi) {  // wait: tile bi loaded AND tile bi-1 stored; then publish
      if (w == 0) __builtin_amdgcn_s_wait_tensorcnt(0);
      __syncthreads();
    }
    // prefetch next tile via TDM into the other buffer (overlaps all compute)
    if (bi < 23 && w == 0)
      tdm_load_2d(OFF_ELD + (buf ^ 1) * 8192,
                  edge_vec + ((size_t)a0 * A_N + (b + 1)) * F_N,
                  64, 16, 64, 16, ROWSTR8);
    if (bi < 23)
      __builtin_prefetch(MB4 + (((size_t)(b + 1) * 8 + w * 2) * 8) * 64, 0, 0);

    // ---- op term via f32 WMMA 16x16x4 (2 f-tiles per wave) ----
    v8f opacc[2];
    for (int j = 0; j < 2; ++j) {
      const int nt = w * 2 + j;
      v8f acc; for (int v = 0; v < 8; ++v) acc[v] = 0.f;
      const float* mb = MB4 + (((size_t)b * 8 + nt) * 8) * 64;  // [kt][lane][2]
      for (int kt = 0; kt < 8; ++kt) {
        v2f bfrag = *(const v2f*)&mb[(kt * 32 + lane) * 2];
        acc = __builtin_amdgcn_wmma_f32_16x16x4_f32(false, la[kt], false, bfrag,
                                                    (short)0, acc, false, false);
      }
      opacc[j] = acc;
    }
    // edge = edge_vec + op + b_out  (each (r,f) written exactly once)
    for (int j = 0; j < 2; ++j) {
      const int f = (w * 2 + j) * 16 + row;
      for (int v = 0; v < 8; ++v) {
        const int r = v + 8 * kh;
        Esm[r][f] = Eld[buf * 2048 + r * 128 + f] + opacc[j][v] + bOutS[f];
      }
    }
    __syncthreads();
    // ---- LayerNorm stats per row ----
    {
      float s = 0.f, q = 0.f;
      for (int f = cc * 16; f < cc * 16 + 16; ++f) { float x = Esm[cr][f]; s += x; q += x * x; }
      pS[cr][cc] = s; pQ[cr][cc] = q;
    }
    __syncthreads();
    if (tid < 16) {
      float s = 0.f, q = 0.f;
      for (int p = 0; p < 8; ++p) { s += pS[tid][p]; q += pQ[tid][p]; }
      float mu = s * (1.0f / F_N);
      muS[tid] = mu;
      rsS[tid] = rsqrtf(q * (1.0f / F_N) - mu * mu + EPSN);
    }
    __syncthreads();
    // ---- bf16 A-fragments of LN(edge) ----
    v16bf a1[4];
    {
      const float mu = muS[row], rs = rsS[row];
      for (int kt = 0; kt < 4; ++kt)
        for (int e = 0; e < 16; ++e) {
          int kl = 16 * kh + e;
          int f = kt * 32 + kl;
          a1[kt][e] = (__bf16)((Esm[row][f] - mu) * rs * lnS[f] + lnB[f]);
        }
    }
    __syncthreads();  // all LN reads of Esm done before GEMM2 atomics
    // ---- GEMM1: relu(LN(E) @ w_t1 + b_t1), 8 N-tiles per wave ----
    for (int j = 0; j < 8; ++j) {
      const int nt = w * 8 + j;
      const int n = nt * 16 + row;
      const float bias = bT1s[n];
      v8f acc; for (int v = 0; v < 8; ++v) acc[v] = bias;
      for (int kt = 0; kt < 4; ++kt) {
        const v16bf bf = *(const v16bf*)&w1s[((kt * 32 + nt) * 32 + lane) * 16];
        acc = __builtin_amdgcn_wmma_f32_16x16x32_bf16(false, a1[kt], false, bf,
                                                      (short)0, acc, false, false);
      }
      for (int v = 0; v < 8; ++v) {
        float x = acc[v] > 0.f ? acc[v] : 0.f;
        t1p[((w * 16) + v + 8 * kh) * 128 + j * 16 + row] = (__bf16)x;
      }
    }
    __syncthreads();
    // ---- A-fragments of this wave's 128-wide hidden slice ----
    v16bf a2[4];
    for (int kt = 0; kt < 4; ++kt)
      for (int e = 0; e < 16; ++e) {
        int kl = 16 * kh + e;
        a2[kt][e] = t1p[((w * 16) + row) * 128 + kt * 32 + kl];
      }
    // ---- GEMM2 partials, reduced into Esm via LDS float atomics ----
    for (int j = 0; j < 8; ++j) {
      v8f acc; for (int v = 0; v < 8; ++v) acc[v] = 0.f;
      for (int kt = 0; kt < 4; ++kt) {
        const int ktg = w * 4 + kt;
        const v16bf bf = *(const v16bf*)&w2s[((ktg * 8 + j) * 32 + lane) * 16];
        acc = __builtin_amdgcn_wmma_f32_16x16x32_bf16(false, a2[kt], false, bf,
                                                      (short)0, acc, false, false);
      }
      const int f = j * 16 + row;
      for (int v = 0; v < 8; ++v)
        atomicAdd(&Esm[v + 8 * kh][f], acc[v]);
    }
    __syncthreads();
    // ---- + b_t2, then async store of the finished tile ----
    for (int q = 0; q < 16; ++q) { const int f = cc * 16 + q; Esm[cr][f] += bT2s[f]; }
    __syncthreads();
    if (w == 0)
      tdm_store_2d(OFF_ESM, out + ((size_t)a0 * A_N + b) * F_N,
                   64, 16, 64, 16, ROWSTR8);
  }
  if (w == 0) __builtin_amdgcn_s_wait_tensorcnt(0);  // drain last store
}

// ---------------------------------------------------------------------------
extern "C" void kernel_launch(void* const* d_in, const int* in_sizes, int n_in,
                              void* d_out, int out_size, void* d_ws, size_t ws_size,
                              hipStream_t stream) {
  const float* node_vec  = (const float*)d_in[0];
  const float* edge_vec  = (const float*)d_in[1];
  const float* node_mask = (const float*)d_in[2];
  // d_in[3] edge_mask: unused by the reference computation
  const float* op_ln_s = (const float*)d_in[4];
  const float* op_ln_b = (const float*)d_in[5];
  const float* w_left  = (const float*)d_in[6];
  const float* b_left  = (const float*)d_in[7];
  const float* w_right = (const float*)d_in[8];
  const float* b_right = (const float*)d_in[9];
  const float* w_out   = (const float*)d_in[10];
  const float* b_out   = (const float*)d_in[11];
  const float* tr_ln_s = (const float*)d_in[12];
  const float* tr_ln_b = (const float*)d_in[13];
  const float* w_t1    = (const float*)d_in[14];
  const float* b_t1    = (const float*)d_in[15];
  const float* w_t2    = (const float*)d_in[16];
  const float* b_t2    = (const float*)d_in[17];
  float* outp = (float*)d_out;

  // workspace carve-up (~6.4 MB)
  float* left   = (float*)d_ws;                 // 384*32
  float* right  = left  + 384 * 32;             // 384*32
  float* MB4    = right + 384 * 32;             // 1572864
  float* leftA4 = MB4   + 1572864;              // 12288
  __bf16* w1B   = (__bf16*)(leftA4 + 12288);    // 65536
  __bf16* w2B   = w1B + 65536;                  // 65536

  prep_act_lr<<<dim3(A_N), dim3(128), 0, stream>>>(node_vec, node_mask, op_ln_s, op_ln_b,
                                                   w_left, b_left, w_right, b_right,
                                                   left, right);
  prep_MB4<<<dim3(6144), dim3(256), 0, stream>>>(right, w_out, MB4);
  prep_leftA4<<<dim3(48), dim3(256), 0, stream>>>(left, leftA4);
  prep_w1B<<<dim3(256), dim3(256), 0, stream>>>(w_t1, w1B);
  prep_w2B<<<dim3(256), dim3(256), 0, stream>>>(w_t2, w2B);

  pair_block_main<<<dim3(16, 24), dim3(128), 0, stream>>>(
      edge_vec, b_out, tr_ln_s, tr_ln_b, b_t1, b_t2,
      leftA4, MB4, w1B, w2B, outp);
}
